// USPAttention_55843164783283
// MI455X (gfx1250) — matrix operand order repaired and checked by reading, not compiled, not run on hardware
//
#include <hip/hip_runtime.h>

typedef __attribute__((ext_vector_type(16))) _Float16 v16h;
typedef __attribute__((ext_vector_type(8)))  _Float16 v8h;
typedef __attribute__((ext_vector_type(8)))  float    v8f;
typedef __attribute__((ext_vector_type(4)))  _Float16 v4h;
typedef __attribute__((ext_vector_type(4)))  float    v4f;

#define B_  2
#define S_  2048
#define H_  16
#define D_  64
#define KT  64             // keys per inner tile (softmax amortized over 64 keys)
#define NT  (S_/KT)        // 32 tiles
#define QW  16             // q rows per wave
#define NWAVE 8
#define QB  (QW*NWAVE)     // 128 q rows per block
#define KSTRIDE  72        // halfs per K-LDS row  (64 d + pad, 144B, 16B aligned)
#define VTSTRIDE 72        // halfs per Vt-LDS row (64 keys + pad, 144B, 16B aligned)
#define PLSTRIDE 72        // halfs per P-LDS row  (64 keys + pad)
#define KBUFSZ  (KT*KSTRIDE)
#define VBUFSZ  (D_*VTSTRIDE)

static __device__ __forceinline__ v16h cat8(v8h lo, v8h hi) {
  return __builtin_shufflevector(lo, hi, 0,1,2,3,4,5,6,7,8,9,10,11,12,13,14,15);
}

// xor-16 lane swap on wave32 as pure VALU (v_permlanex16_b32, identity selects)
static __device__ __forceinline__ float xswap16(float x) {
  int xi = __builtin_bit_cast(int, x);
  int r  = __builtin_amdgcn_permlanex16(xi, xi, 0x76543210, 0xfedcba98, false, false);
  return __builtin_bit_cast(float, r);
}

__global__ __launch_bounds__(256)
void usp_attn_fwd(const float* __restrict__ Q,
                  const float* __restrict__ K,
                  const float* __restrict__ V,
                  float* __restrict__ O) {
  __shared__ __align__(16) _Float16 Klds[2 * KBUFSZ];            // K tile [key][d], double buffered
  __shared__ __align__(16) _Float16 Vt[2 * VBUFSZ];              // V tile transposed [d][key]
  __shared__ __align__(16) _Float16 Plds[NWAVE * QW * PLSTRIDE]; // per-wave P [q][key 0..63]

  const int t    = threadIdx.x;
  const int wave = t >> 5;
  const int lane = t & 31;
  const int lmod = lane & 15;
  const int hsel = lane >> 4;

  const int qblk = blockIdx.x & 15;        // S_/QB == 16
  const int bh   = blockIdx.x >> 4;
  const int b    = bh >> 4;                // H_ == 16
  const int h    = bh & 15;
  const int q0   = qblk * QB + wave * QW;

  // ---- Resident Q B-fragments, with scale*log2(e) folded in (base-2 softmax)
  const float qs = 0.125f * 1.44269504088896f;
  v16h bq0, bq1;
  {
    const float* qrow = Q + (((size_t)b * S_ + (q0 + lmod)) * H_ + h) * D_;
#pragma unroll
    for (int e = 0; e < 16; ++e) {
      bq0[e] = (_Float16)(qrow[8 * hsel + e + (e >= 8 ? 8 : 0)] * qs);
      bq1[e] = (_Float16)(qrow[32 + 8 * hsel + e + (e >= 8 ? 8 : 0)] * qs);
    }
  }

  v8f oacc[4];
#pragma unroll
  for (int nc = 0; nc < 4; ++nc) oacc[nc] = (v8f){};
  float mrun = -__builtin_inff();
  float lrun = 0.0f;

  // ---- Global->register staging: each thread owns a 4(keys)x4(d) fp32 block
  const int ss  = t >> 4;           // 0..15 -> key rows 4*ss..4*ss+3
  const int sc4 = t & 15;           // d-group of 4
  v4f kbuf[4], vbuf[4];

  auto loadregs = [&](int tile) {
#pragma unroll
    for (int i = 0; i < 4; ++i) {
      const size_t g = (((size_t)b * S_ + ((size_t)tile * KT + 4 * ss + i)) * H_ + h) * D_ + sc4 * 4;
      kbuf[i] = *(const v4f*)(K + g);
      vbuf[i] = *(const v4f*)(V + g);
    }
  };
  auto stage = [&](int bufi) {
    _Float16* kb = &Klds[bufi * KBUFSZ];
    _Float16* vb = &Vt[bufi * VBUFSZ];
#pragma unroll
    for (int i = 0; i < 4; ++i) {     // K: row-major, 4 x b64 stores
      v4h kh;
#pragma unroll
      for (int j = 0; j < 4; ++j) kh[j] = (_Float16)kbuf[i][j];
      *(v4h*)(&kb[(4 * ss + i) * KSTRIDE + sc4 * 4]) = kh;
    }
#pragma unroll
    for (int j = 0; j < 4; ++j) {     // V: 4x4 register transpose, 4 x b64 stores
      v4h w;
#pragma unroll
      for (int i = 0; i < 4; ++i) w[i] = (_Float16)vbuf[i][j];
      *(v4h*)(&vb[(4 * sc4 + j) * VTSTRIDE + 4 * ss]) = w;
    }
  };

  loadregs(0);
  stage(0);
  loadregs(1);

  for (int it = 0; it < NT; ++it) {
    const int cur = it & 1;
    __syncthreads();  // buf[cur] staged; all waves done reading buf[cur^1]

    if (it + 1 < NT) {
      stage(cur ^ 1);                 // overlaps this tile's compute
      if (it + 2 < NT) loadregs(it + 2);
    }

    const _Float16* kb = &Klds[cur * KBUFSZ];
    const _Float16* vb = &Vt[cur * VBUFSZ];

    // ---- S^T = K * Q^T : 4 key-chunks x 2 chained d-halves = 8 WMMAs
    v8f st[4];
    const v8f cz = (v8f){};
#pragma unroll
    for (int mc = 0; mc < 4; ++mc) {
      const _Float16* kr = &kb[(mc * 16 + lmod) * KSTRIDE + 8 * hsel];
      v16h alo = cat8(*(const v8h*)(kr),      *(const v8h*)(kr + 16));
      v16h ahi = cat8(*(const v8h*)(kr + 32), *(const v8h*)(kr + 48));
      st[mc] = __builtin_amdgcn_wmma_f32_16x16x32_f16(false, alo, false, bq0, (short)0, cz,     false, false);
      st[mc] = __builtin_amdgcn_wmma_f32_16x16x32_f16(false, ahi, false, bq1, (short)0, st[mc], false, false);
    }
    // Per lane: st[mc][r] = s2(key mc*16 + r + 8*hsel, q=lmod)

    // ---- ONE online-softmax update per 64 keys (per-lane scalar stats)
    float pm[16];
#pragma unroll
    for (int r = 0; r < 8; ++r) {
      pm[r]     = fmaxf(st[0][r], st[1][r]);
      pm[8 + r] = fmaxf(st[2][r], st[3][r]);
    }
#pragma unroll
    for (int s = 8; s >= 1; s >>= 1)
#pragma unroll
      for (int r = 0; r < s; ++r) pm[r] = fmaxf(pm[r], pm[r + s]);
    const float vmax  = fmaxf(pm[0], xswap16(pm[0]));
    const float mnew  = fmaxf(mrun, vmax);
    const float alpha = __builtin_amdgcn_exp2f(mrun - mnew);
    float p[4][8];
#pragma unroll
    for (int mc = 0; mc < 4; ++mc)
#pragma unroll
      for (int r = 0; r < 8; ++r)
        p[mc][r] = __builtin_amdgcn_exp2f(st[mc][r] - mnew);
    float ts[16];
#pragma unroll
    for (int r = 0; r < 8; ++r) {
      ts[r]     = p[0][r] + p[1][r];
      ts[8 + r] = p[2][r] + p[3][r];
    }
#pragma unroll
    for (int s = 8; s >= 1; s >>= 1)
#pragma unroll
      for (int r = 0; r < s; ++r) ts[r] += ts[r + s];
    const float tsum = ts[0] + xswap16(ts[0]);
    lrun = alpha * lrun + tsum;
    mrun = mnew;
#pragma unroll
    for (int nc = 0; nc < 4; ++nc)
#pragma unroll
      for (int r = 0; r < 8; ++r) oacc[nc][r] *= alpha;

    // ---- Stage P[q][key 0..63]: 4 contiguous b128 stores per lane
    _Float16* pl = &Plds[wave * QW * PLSTRIDE];
#pragma unroll
    for (int mc = 0; mc < 4; ++mc) {
      v8h ph;
#pragma unroll
      for (int r = 0; r < 8; ++r) ph[r] = (_Float16)p[mc][r];
      *(v8h*)(&pl[lmod * PLSTRIDE + mc * 16 + 8 * hsel]) = ph;
    }
    // Same-wave LDS store->load ordering only (wave-private region; LDS in-order)
    __builtin_amdgcn_fence(__ATOMIC_ACQ_REL, "workgroup");
    __builtin_amdgcn_wave_barrier();

    // ---- O^T += V^T * P^T : 4 d-chunks x 2 chained key-halves = 8 WMMAs
    const _Float16* pr = &pl[lmod * PLSTRIDE + 8 * hsel];
    v16h bp0 = cat8(*(const v8h*)(pr),      *(const v8h*)(pr + 16));
    v16h bp1 = cat8(*(const v8h*)(pr + 32), *(const v8h*)(pr + 48));
#pragma unroll
    for (int nc = 0; nc < 4; ++nc) {
      const _Float16* vr = &vb[(nc * 16 + lmod) * VTSTRIDE + 8 * hsel];
      v16h av0 = cat8(*(const v8h*)(vr),      *(const v8h*)(vr + 16));
      v16h av1 = cat8(*(const v8h*)(vr + 32), *(const v8h*)(vr + 48));
      oacc[nc] = __builtin_amdgcn_wmma_f32_16x16x32_f16(false, av0, false, bp0, (short)0, oacc[nc], false, false);
      oacc[nc] = __builtin_amdgcn_wmma_f32_16x16x32_f16(false, av1, false, bp1, (short)0, oacc[nc], false, false);
    }
  }

  // ---- Epilogue: lane holds 8 consecutive d per fragment -> vector stores
  const float invl = 1.0f / lrun;
  float* orow = O + (((size_t)b * S_ + (q0 + lmod)) * H_ + h) * D_;
#pragma unroll
  for (int nc = 0; nc < 4; ++nc) {
    v4f o0, o1;
#pragma unroll
    for (int r = 0; r < 4; ++r) { o0[r] = oacc[nc][r] * invl; o1[r] = oacc[nc][r + 4] * invl; }
    *(v4f*)(&orow[nc * 16 + 8 * hsel])     = o0;
    *(v4f*)(&orow[nc * 16 + 8 * hsel + 4]) = o1;
  }
}

extern "C" void kernel_launch(void* const* d_in, const int* in_sizes, int n_in,
                              void* d_out, int out_size, void* d_ws, size_t ws_size,
                              hipStream_t stream) {
  (void)in_sizes; (void)n_in; (void)out_size; (void)d_ws; (void)ws_size;
  const float* q = (const float*)d_in[0];
  const float* k = (const float*)d_in[1];
  const float* v = (const float*)d_in[2];
  float* o = (float*)d_out;
  dim3 grid(B_ * H_ * (S_ / QB));  // 512 blocks
  dim3 block(256);                 // 8 wave32 waves
  usp_attn_fwd<<<grid, block, 0, stream>>>(q, k, v, o);
}
